// Super_MixMamba_43181601194405
// MI455X (gfx1250) — matrix operand order repaired and checked by reading, not compiled, not run on hardware
//
#include <hip/hip_runtime.h>
#include <hip/hip_bf16.h>
#include <math.h>

typedef __bf16 bf16;
typedef __attribute__((ext_vector_type(16))) __bf16 v16bf;
typedef __attribute__((ext_vector_type(8)))  __bf16 v8bf;
typedef __attribute__((ext_vector_type(8)))  float  v8f;

// ---- static config (matches reference) ----
constexpr int B_SZ  = 8;
constexpr int H_DIM = 24, W_DIM = 24;
constexpr int L     = H_DIM * W_DIM;     // 576
constexpr int D_EMB = 384;
constexpr int D_IN  = 768;               // 2 * D_EMB
constexpr int N_ST  = 16;
constexpr int R_RNK = 24;                // dt_rank
constexpr int KCONV = 4;
constexpr int NDIR  = 4;
constexpr int RC    = 96;
constexpr int M_TOK = B_SZ * L;          // 4608 tokens
constexpr float LN_EPS = 1e-5f;

__device__ __forceinline__ int perm_idx(int dir, int t) {
    if (dir == 0) return t;
    if (dir == 1) return L - 1 - t;
    int u = (dir == 2) ? t : (L - 1 - t);
    return (u % H_DIM) * W_DIM + (u / H_DIM);
}

// ---------------------------------------------------------------------------
// bf16 WMMA GEMM:  C[M,Nn] = A[M,K] * Bw[Nn,K]^T   (both K-major)
// Each wave computes a 16x64 strip: 4 accumulators share one A fragment per
// K-step (4 independent WMMAs -> XDL latency hiding, A refetch cut 4x).
// Requires Nn % 64 == 0, M % 16 == 0, K % 32 == 0 (true for all calls here).
// Writes either fp32 (Cf) or bf16 (Cb).
// ---------------------------------------------------------------------------
__global__ __launch_bounds__(128) void wmma_gemm_bf16(
    const bf16* __restrict__ A, int lda,
    const bf16* __restrict__ Bw, int ldb,
    float* __restrict__ Cf, bf16* __restrict__ Cb, int ldc,
    int M, int Nn, int K)
{
    const int lane = threadIdx.x & 31;
    const int wave = threadIdx.x >> 5;
    const int nstrips = Nn >> 6;                  // strips of 4 N-tiles
    const int strip = blockIdx.x * 4 + wave;
    if (strip >= (M >> 4) * nstrips) return;      // wave-uniform: EXEC all-ones
    const int tm  = strip / nstrips;
    const int sn  = strip - tm * nstrips;         // strip index along N
    const int half = lane >> 4;
    const int l15  = lane & 15;

    // A fragment: row = tm*16 + l15; lane<16 holds K 0..15, lane>=16 K 16..31
    const bf16* ap = A + (size_t)(tm * 16 + l15) * lda + half * 16;
    // B fragments: col j*16 + l15 within strip; lane<16 holds K 0..15, lane>=16 K 16..31
    const bf16* bp = Bw + (size_t)(sn * 64 + l15) * ldb + half * 16;
    const size_t bstep = (size_t)16 * ldb;        // next 16 output columns

    v8f acc0 = {}, acc1 = {}, acc2 = {}, acc3 = {};
    for (int k0 = 0; k0 < K; k0 += 32) {
        v8bf alo = *(const v8bf*)(ap + k0);
        v8bf ahi = *(const v8bf*)(ap + k0 + 8);
        v16bf a;
#pragma unroll
        for (int i = 0; i < 8; ++i) { a[i] = alo[i]; a[i + 8] = ahi[i]; }
        v16bf b0 = *(const v16bf*)(bp + k0);
        v16bf b1 = *(const v16bf*)(bp + bstep + k0);
        v16bf b2 = *(const v16bf*)(bp + 2 * bstep + k0);
        v16bf b3 = *(const v16bf*)(bp + 3 * bstep + k0);
        if (k0 + 32 < K) {
            __builtin_prefetch(ap + k0 + 32, 0, 1);          // global_prefetch_b8
            __builtin_prefetch(bp + k0 + 32, 0, 1);
            __builtin_prefetch(bp + 3 * bstep + k0 + 32, 0, 1);
        }
        // 4 independent WMMAs sharing the A fragment (no RAW between them)
        acc0 = __builtin_amdgcn_wmma_f32_16x16x32_bf16(false, a, false, b0, (short)0, acc0, false, false);
        acc1 = __builtin_amdgcn_wmma_f32_16x16x32_bf16(false, a, false, b1, (short)0, acc1, false, false);
        acc2 = __builtin_amdgcn_wmma_f32_16x16x32_bf16(false, a, false, b2, (short)0, acc2, false, false);
        acc3 = __builtin_amdgcn_wmma_f32_16x16x32_bf16(false, a, false, b3, (short)0, acc3, false, false);
    }

    const int m0 = tm * 16 + half * 8;            // VGPR v -> M = m0 + v
    const int n0 = sn * 64 + l15;
    if (Cf) {
#pragma unroll
        for (int v = 0; v < 8; ++v) {
            float* cr = Cf + (size_t)(m0 + v) * ldc + n0;
            cr[0]  = acc0[v];
            cr[16] = acc1[v];
            cr[32] = acc2[v];
            cr[48] = acc3[v];
        }
    } else {
#pragma unroll
        for (int v = 0; v < 8; ++v) {
            bf16* cr = Cb + (size_t)(m0 + v) * ldc + n0;
            cr[0]  = (bf16)acc0[v];
            cr[16] = (bf16)acc1[v];
            cr[32] = (bf16)acc2[v];
            cr[48] = (bf16)acc3[v];
        }
    }
}

// ---------------------------------------------------------------------------
// Elementwise casts / packs
// ---------------------------------------------------------------------------
__global__ void cast_f32_bf16(const float* __restrict__ src, bf16* __restrict__ dst, int n)
{
    int i = blockIdx.x * blockDim.x + threadIdx.x;
    if (i < n) dst[i] = (bf16)src[i];
}

// xproj_ws (4,56,768) -> bf16 padded (4,64,768), rows 56..63 = 0
__global__ void pad_xproj(const float* __restrict__ src, bf16* __restrict__ dst)
{
    int i = blockIdx.x * blockDim.x + threadIdx.x;
    if (i >= NDIR * 64 * D_IN) return;
    int d   = i % D_IN;
    int r   = (i / D_IN) % 64;
    int dir = i / (64 * D_IN);
    dst[i] = (r < R_RNK + 2 * N_ST)
           ? (bf16)src[((size_t)dir * (R_RNK + 2 * N_ST) + r) * D_IN + d]
           : (bf16)0.f;
}

// dt_ws (4,768,24) -> bf16 padded (4,768,32), cols 24..31 = 0
__global__ void pad_dtw(const float* __restrict__ src, bf16* __restrict__ dst)
{
    int i = blockIdx.x * blockDim.x + threadIdx.x;
    if (i >= NDIR * D_IN * 32) return;
    int r  = i % 32;
    int dd = i / 32;                      // dir*768 + d
    dst[i] = (r < R_RNK) ? (bf16)src[(size_t)dd * R_RNK + r] : (bf16)0.f;
}

// dtr = xdbl[:, :, 0:24] -> bf16 padded to 32 cols
__global__ void extract_dtr(const float* __restrict__ xdbl, bf16* __restrict__ dtr)
{
    int i = blockIdx.x * blockDim.x + threadIdx.x;
    if (i >= NDIR * M_TOK * 32) return;
    int r = i % 32;
    size_t tok = (size_t)(i / 32);
    dtr[i] = (r < R_RNK) ? (bf16)xdbl[tok * 64 + r] : (bf16)0.f;
}

// ---------------------------------------------------------------------------
// Depthwise causal conv (K=4) + SiLU, in per-direction scan order.
// One thread per (dir, b, d) channel; sliding window in registers.
// ---------------------------------------------------------------------------
__global__ void conv_silu_kernel(const float* __restrict__ xz,
                                 const float* __restrict__ cw,
                                 const float* __restrict__ cb,
                                 bf16* __restrict__ xconv)
{
    int gid = blockIdx.x * blockDim.x + threadIdx.x;
    if (gid >= NDIR * B_SZ * D_IN) return;
    int d   = gid % D_IN;
    int rem = gid / D_IN;
    int b   = rem % B_SZ;
    int dir = rem / B_SZ;

    const float* w = cw + (size_t)(dir * D_IN + d) * KCONV;
    float w0 = w[0], w1 = w[1], w2 = w[2], w3 = w[3];
    float bias = cb[dir * D_IN + d];

    const float* xin = xz + (size_t)b * L * (2 * D_IN) + d;   // x-half of xz
    bf16* out = xconv + (size_t)(dir * B_SZ + b) * L * D_IN + d;

    float x0 = 0.f, x1 = 0.f, x2 = 0.f;
    for (int t = 0; t < L; ++t) {
        int p = perm_idx(dir, t);
        float x3 = xin[(size_t)p * (2 * D_IN)];
        float a  = w0 * x0 + w1 * x1 + w2 * x2 + w3 * x3 + bias;
        float s  = a / (1.f + expf(-a));                      // silu
        out[(size_t)t * D_IN] = (bf16)s;
        x0 = x1; x1 = x2; x2 = x3;
    }
}

// ---------------------------------------------------------------------------
// Selective scan. Block = (dir, b); 768 threads = one channel each.
// B/C state vectors (32 floats/step) staged through LDS.
// Fuses softplus(delta), +u*D, silu(z) gate, and the inverse permute on store.
// ---------------------------------------------------------------------------
__global__ __launch_bounds__(D_IN) void scan_kernel(
    const bf16* __restrict__ xconv, const bf16* __restrict__ delta_raw,
    const float* __restrict__ xdbl, const float* __restrict__ xz,
    const float* __restrict__ A_logs, const float* __restrict__ dt_bs,
    const float* __restrict__ Dsv, bf16* __restrict__ ys)
{
    __shared__ float sBC[2 * N_ST];
    const int db  = blockIdx.x;                 // dir*B + b
    const int dir = db / B_SZ;
    const int b   = db % B_SZ;
    const int d   = threadIdx.x;

    float Arow[N_ST], h[N_ST];
#pragma unroll
    for (int n = 0; n < N_ST; ++n) {
        Arow[n] = -expf(A_logs[((size_t)dir * D_IN + d) * N_ST + n]);
        h[n] = 0.f;
    }
    const float dtb = dt_bs[dir * D_IN + d];
    const float Dp  = Dsv[dir * D_IN + d];

    const bf16*  up = xconv     + (size_t)db * L * D_IN + d;
    const bf16*  dp = delta_raw + (size_t)db * L * D_IN + d;
    const float* bc = xdbl      + (size_t)db * L * 64;
    const float* zp = xz        + (size_t)b * L * (2 * D_IN) + D_IN + d;
    bf16* yp = ys + (size_t)db * L * D_IN + d;

    for (int t = 0; t < L; ++t) {
        if (threadIdx.x < 2 * N_ST)
            sBC[threadIdx.x] = bc[(size_t)t * 64 + R_RNK + threadIdx.x];
        __syncthreads();

        float dr    = (float)dp[(size_t)t * D_IN] + dtb;
        float delta = dr > 20.f ? dr : log1pf(expf(dr));      // softplus
        float u     = (float)up[(size_t)t * D_IN];
        float du    = delta * u;
        float y     = u * Dp;
#pragma unroll
        for (int n = 0; n < N_ST; ++n) {
            h[n] = h[n] * __expf(delta * Arow[n]) + du * sBC[n];
            y   += h[n] * sBC[N_ST + n];
        }
        int p   = perm_idx(dir, t);                           // inverse permute on store
        float z = zp[(size_t)p * (2 * D_IN)];
        y      *= z / (1.f + expf(-z));                       // * silu(z)
        yp[(size_t)p * D_IN] = (bf16)y;
        __syncthreads();
    }
}

// ---------------------------------------------------------------------------
// BiAttn: per-token layernorm stats, channel-mean of normalized ys,
// tiny MLP gate, apply gate + sum over directions.
// ---------------------------------------------------------------------------
__global__ __launch_bounds__(256) void ln_stats_kernel(
    const bf16* __restrict__ ys, float* __restrict__ mustats)
{
    __shared__ float ss[256], sq[256];
    size_t tok = blockIdx.x;                    // over NDIR*B*L
    const bf16* yp = ys + tok * D_IN;
    float s = 0.f, q = 0.f;
    for (int d = threadIdx.x; d < D_IN; d += 256) {
        float v = (float)yp[d]; s += v; q += v * v;
    }
    ss[threadIdx.x] = s; sq[threadIdx.x] = q;
    __syncthreads();
    for (int st = 128; st > 0; st >>= 1) {
        if (threadIdx.x < st) { ss[threadIdx.x] += ss[threadIdx.x + st];
                                sq[threadIdx.x] += sq[threadIdx.x + st]; }
        __syncthreads();
    }
    if (threadIdx.x == 0) {
        float mu  = ss[0] / D_IN;
        float var = sq[0] / D_IN - mu * mu;
        mustats[tok * 2]     = mu;
        mustats[tok * 2 + 1] = rsqrtf(var + LN_EPS);
    }
}

__global__ void bi_accum_kernel(const bf16* __restrict__ ys,
                                const float* __restrict__ mustats,
                                const float* __restrict__ gamma,
                                const float* __restrict__ beta,
                                float* __restrict__ xnmean)
{
    int gid = blockIdx.x * blockDim.x + threadIdx.x;
    if (gid >= NDIR * B_SZ * D_IN) return;
    int d  = gid % D_IN;
    int db = gid / D_IN;
    const bf16*  yp = ys + (size_t)db * L * D_IN + d;
    const float* ms = mustats + (size_t)db * L * 2;
    float acc = 0.f;
    for (int t = 0; t < L; ++t)
        acc += ((float)yp[(size_t)t * D_IN] - ms[t * 2]) * ms[t * 2 + 1];
    xnmean[gid] = (acc / L) * gamma[d] + beta[d];
}

__global__ __launch_bounds__(128) void gate_kernel(
    const float* __restrict__ xnmean,
    const float* __restrict__ rw, const float* __restrict__ rb,
    const float* __restrict__ sw, const float* __restrict__ sb,
    float* __restrict__ gate)
{
    __shared__ float g[RC];
    int db  = blockIdx.x;
    int tid = threadIdx.x;
    const float* xm = xnmean + (size_t)db * D_IN;
    if (tid < RC) {
        float s = rb[tid];
        const float* w = rw + (size_t)tid * D_IN;
        for (int d = 0; d < D_IN; ++d) s += xm[d] * w[d];
        float c = 0.7978845608028654f * (s + 0.044715f * s * s * s);
        g[tid] = 0.5f * s * (1.f + tanhf(c));                 // gelu (tanh approx)
    }
    __syncthreads();
    for (int d = tid; d < D_IN; d += 128) {
        float s = sb[d];
        const float* w = sw + (size_t)d * RC;
        for (int r = 0; r < RC; ++r) s += g[r] * w[r];
        gate[(size_t)db * D_IN + d] = 1.f / (1.f + expf(-s)); // sigmoid
    }
}

__global__ void apply_gate_kernel(const bf16* __restrict__ ys,
                                  const float* __restrict__ gate,
                                  bf16* __restrict__ ysum)
{
    int i = blockIdx.x * blockDim.x + threadIdx.x;
    if (i >= M_TOK * D_IN) return;
    int d  = i % D_IN;
    int bt = i / D_IN;
    int b  = bt / L;
    float s = 0.f;
#pragma unroll
    for (int dir = 0; dir < NDIR; ++dir)
        s += (float)ys[(size_t)dir * M_TOK * D_IN + i]
           * gate[(size_t)(dir * B_SZ + b) * D_IN + d];
    ysum[i] = (bf16)s;
}

// ---------------------------------------------------------------------------
extern "C" void kernel_launch(void* const* d_in, const int* in_sizes, int n_in,
                              void* d_out, int out_size, void* d_ws, size_t ws_size,
                              hipStream_t stream)
{
    const float* hs     = (const float*)d_in[0];
    const float* w_in   = (const float*)d_in[1];   // (1536, 384)
    const float* w_out  = (const float*)d_in[2];   // (384, 768)
    const float* A_logs = (const float*)d_in[3];   // (4, 768, 16)
    const float* conv_w = (const float*)d_in[4];   // (4, 768, 4)
    const float* conv_b = (const float*)d_in[5];   // (4, 768)
    const float* xproj  = (const float*)d_in[6];   // (4, 56, 768)
    const float* dt_w   = (const float*)d_in[7];   // (4, 768, 24)
    const float* dt_b   = (const float*)d_in[8];   // (4, 768)
    const float* Dsv    = (const float*)d_in[9];   // (4, 768)
    const float* gamma  = (const float*)d_in[10];
    const float* beta   = (const float*)d_in[11];
    const float* rw     = (const float*)d_in[12];  // (96, 768)
    const float* rb     = (const float*)d_in[13];
    const float* sw     = (const float*)d_in[14];  // (768, 96)
    const float* sb     = (const float*)d_in[15];
    float* out = (float*)d_out;                    // (B, L, 384) fp32

    // ---- workspace sub-allocation (256B aligned) ----
    char* ws = (char*)d_ws;
    size_t off = 0;
    auto alloc = [&](size_t bytes) -> char* {
        char* p = ws + off;
        off += (bytes + 255) & ~(size_t)255;
        return p;
    };
    bf16*  hs_bf    = (bf16*) alloc((size_t)M_TOK * D_EMB * 2);       // 3.5 MB
    bf16*  win_bf   = (bf16*) alloc((size_t)2 * D_IN * D_EMB * 2);    // 1.2 MB
    bf16*  wout_bf  = (bf16*) alloc((size_t)D_EMB * D_IN * 2);        // 0.6 MB
    bf16*  wxp_bf   = (bf16*) alloc((size_t)NDIR * 64 * D_IN * 2);    // 0.4 MB
    bf16*  wdt_bf   = (bf16*) alloc((size_t)NDIR * D_IN * 32 * 2);    // 0.2 MB
    float* xz_f     = (float*)alloc((size_t)M_TOK * 2 * D_IN * 4);    // 28.3 MB
    bf16*  xconv_bf = (bf16*) alloc((size_t)NDIR * M_TOK * D_IN * 2); // 28.3 MB
    float* xdbl_f   = (float*)alloc((size_t)NDIR * M_TOK * 64 * 4);   // 4.7 MB
    bf16*  dtr_bf   = (bf16*) alloc((size_t)NDIR * M_TOK * 32 * 2);   // 1.2 MB
    bf16*  delta_bf = (bf16*) alloc((size_t)NDIR * M_TOK * D_IN * 2); // 28.3 MB
    bf16*  ys_bf    = (bf16*) alloc((size_t)NDIR * M_TOK * D_IN * 2); // 28.3 MB
    float* mustats  = (float*)alloc((size_t)NDIR * M_TOK * 2 * 4);    // 0.15 MB
    float* xnmean   = (float*)alloc((size_t)NDIR * B_SZ * D_IN * 4);
    float* gatebuf  = (float*)alloc((size_t)NDIR * B_SZ * D_IN * 4);
    bf16*  ysum_bf  = (bf16*) alloc((size_t)M_TOK * D_IN * 2);        // 7.1 MB
    (void)ws_size; (void)in_sizes; (void)n_in; (void)out_size;

    auto cdiv = [](int a, int b) { return (a + b - 1) / b; };
    auto gemm = [&](const bf16* A, int lda, const bf16* Bw, int ldb,
                    float* Cf, bf16* Cb, int ldc, int M, int Nn, int K) {
        int strips = (M / 16) * (Nn / 64);
        wmma_gemm_bf16<<<dim3(cdiv(strips, 4)), dim3(128), 0, stream>>>(
            A, lda, Bw, ldb, Cf, Cb, ldc, M, Nn, K);
    };

    // 1) casts / weight packing
    { int n = M_TOK * D_EMB;
      cast_f32_bf16<<<cdiv(n,256),256,0,stream>>>(hs, hs_bf, n); }
    { int n = 2 * D_IN * D_EMB;
      cast_f32_bf16<<<cdiv(n,256),256,0,stream>>>(w_in, win_bf, n); }
    { int n = D_EMB * D_IN;
      cast_f32_bf16<<<cdiv(n,256),256,0,stream>>>(w_out, wout_bf, n); }
    { int n = NDIR * 64 * D_IN;
      pad_xproj<<<cdiv(n,256),256,0,stream>>>(xproj, wxp_bf); }
    { int n = NDIR * D_IN * 32;
      pad_dtw<<<cdiv(n,256),256,0,stream>>>(dt_w, wdt_bf); }

    // 2) in_proj: xz (4608,1536) = hs (4608,384) x w_in^T
    gemm(hs_bf, D_EMB, win_bf, D_EMB, xz_f, nullptr, 2 * D_IN,
         M_TOK, 2 * D_IN, D_EMB);

    // 3) depthwise causal conv + silu (per-direction scan order)
    { int n = NDIR * B_SZ * D_IN;
      conv_silu_kernel<<<cdiv(n,256),256,0,stream>>>(xz_f, conv_w, conv_b, xconv_bf); }

    // 4) x-proj per direction: xdbl (4608,64) = xconv (4608,768) x wxp^T
    for (int dir = 0; dir < NDIR; ++dir)
        gemm(xconv_bf + (size_t)dir * M_TOK * D_IN, D_IN,
             wxp_bf   + (size_t)dir * 64 * D_IN,    D_IN,
             xdbl_f   + (size_t)dir * M_TOK * 64, nullptr, 64,
             M_TOK, 64, D_IN);

    // 5) dtr extraction (pad K 24 -> 32)
    { int n = NDIR * M_TOK * 32;
      extract_dtr<<<cdiv(n,256),256,0,stream>>>(xdbl_f, dtr_bf); }

    // 6) delta projection per direction: (4608,768) = dtr (4608,32) x dt_w^T
    for (int dir = 0; dir < NDIR; ++dir)
        gemm(dtr_bf   + (size_t)dir * M_TOK * 32, 32,
             wdt_bf   + (size_t)dir * D_IN * 32,  32,
             nullptr, delta_bf + (size_t)dir * M_TOK * D_IN, D_IN,
             M_TOK, D_IN, 32);

    // 7) selective scan (+softplus, +u*D, *silu(z), inverse-permute on store)
    scan_kernel<<<dim3(NDIR * B_SZ), dim3(D_IN), 0, stream>>>(
        xconv_bf, delta_bf, xdbl_f, xz_f, A_logs, dt_b, Dsv, ys_bf);

    // 8) BiAttn gate
    ln_stats_kernel<<<dim3(NDIR * M_TOK), dim3(256), 0, stream>>>(ys_bf, mustats);
    { int n = NDIR * B_SZ * D_IN;
      bi_accum_kernel<<<cdiv(n,256),256,0,stream>>>(ys_bf, mustats, gamma, beta, xnmean); }
    gate_kernel<<<dim3(NDIR * B_SZ), dim3(128), 0, stream>>>(xnmean, rw, rb, sw, sb, gatebuf);
    { int n = M_TOK * D_IN;
      apply_gate_kernel<<<cdiv(n,256),256,0,stream>>>(ys_bf, gatebuf, ysum_bf); }

    // 9) out_proj: out (4608,384) = ysum (4608,768) x w_out^T
    gemm(ysum_bf, D_IN, wout_bf, D_IN, out, nullptr, D_EMB,
         M_TOK, D_EMB, D_IN);
}
